// AttentionLayer_85117661872916
// MI455X (gfx1250) — matrix operand order repaired and checked by reading, not compiled, not run on hardware
//
#include <hip/hip_runtime.h>

#define DEV __device__ __forceinline__

typedef __attribute__((ext_vector_type(16))) __bf16 v16bf;
typedef __attribute__((ext_vector_type(8)))  __bf16 bf16x8;
typedef __attribute__((ext_vector_type(8)))  float  v8f;

static constexpr int D_MODEL  = 1024;
static constexpr int N_HEADS  = 16;
static constexpr int HEAD_DIM = 64;
static constexpr int BATCH    = 2;
static constexpr int SEQ      = 2048;

DEV v8f wmma_bf16(v16bf a, v16bf b, v8f c) {
  return __builtin_amdgcn_wmma_f32_16x16x32_bf16(false, a, false, b, (short)0, c,
                                                 false, false);
}

// A-matrix fragment: 16x32 bf16 row of M=lane%16.
// lanes 0-15 hold K 0..7 (v0-3) and 16..23 (v4-7); lanes 16-31 hold K 8..15, 24..31.
DEV v16bf ld_frag_a(const __bf16* row, int half) {
  union { v16bf v; bf16x8 h[2]; } u;
  u.h[0] = *(const bf16x8*)(row + half * 8);
  u.h[1] = *(const bf16x8*)(row + 16 + half * 8);
  return u.v;
}

// B-matrix fragment: 32x16 bf16, source row = column N (contiguous contraction-K).
// lanes 0-15 hold K 0..7, 16..23; lanes 16-31 hold K 8..15, 24..31.
DEV v16bf ld_frag_b(const __bf16* col, int half) {
  union { v16bf v; bf16x8 h[2]; } u;
  u.h[0] = *(const bf16x8*)(col + half * 8);
  u.h[1] = *(const bf16x8*)(col + 16 + half * 8);
  return u.v;
}

// Async global -> LDS copy of 16 bytes (CDNA5 ASYNCcnt path).
// LDS generic addresses truncate to the LDS byte offset in the low 32 bits.
DEV void async_copy16(const __bf16* gsrc, __bf16* ldst) {
  unsigned lds_off = (unsigned)(size_t)ldst;
  asm volatile("global_load_async_to_lds_b128 %0, %1, off"
               :: "v"(lds_off), "v"(gsrc) : "memory");
}
// Wait until at most the N most recently issued async copies are outstanding
// (async loads complete in order, so this drains all older tiles).
DEV void async_wait4() { asm volatile("s_wait_asynccnt 4" ::: "memory"); }
DEV void async_wait5() { asm volatile("s_wait_asynccnt 5" ::: "memory"); }
DEV void async_wait0() { asm volatile("s_wait_asynccnt 0" ::: "memory"); }

// ---------------------------------------------------------------------------
// Kernel 1: transpose + convert weights: WT[o][i] = (bf16) W[i][o]
// ---------------------------------------------------------------------------
__global__ __launch_bounds__(256) void wtrans_kernel(
    const float* Wq, const float* Wk, const float* Wv, const float* Wo,
    __bf16* WqT, __bf16* WkT, __bf16* WvT, __bf16* WoT) {
  const float* src = (blockIdx.z == 0) ? Wq : (blockIdx.z == 1) ? Wk
                    : (blockIdx.z == 2) ? Wv : Wo;
  __bf16* dst = (blockIdx.z == 0) ? WqT : (blockIdx.z == 1) ? WkT
               : (blockIdx.z == 2) ? WvT : WoT;
  __shared__ float tile[32][33];
  int x = blockIdx.x * 32 + threadIdx.x;
  int y = blockIdx.y * 32 + threadIdx.y;
#pragma unroll
  for (int i = 0; i < 32; i += 8)
    tile[threadIdx.y + i][threadIdx.x] = src[(size_t)(y + i) * D_MODEL + x];
  __syncthreads();
  int x2 = blockIdx.y * 32 + threadIdx.x;
  int y2 = blockIdx.x * 32 + threadIdx.y;
#pragma unroll
  for (int i = 0; i < 32; i += 8)
    dst[(size_t)(y2 + i) * D_MODEL + x2] = (__bf16)tile[threadIdx.x][threadIdx.y + i];
}

// ---------------------------------------------------------------------------
// Kernel 2: fused QKV projection.  Y = X @ W + b, block tile 64(M) x 256(N),
// 8 waves as 2x4, each wave 32x64 -> 8 WMMAs per K-step of 32.
// B tile: double-buffered global_load_async_to_lds_b128 (copies for tile t+1
// overlap the WMMAs of tile t; s_wait_asynccnt 4 drains only older tiles).
// blockIdx.z: 0 -> Q[b][h][l][dh], 1 -> K[b][h][l][dh], 2 -> Vt[b][h][dh][l]
// ---------------------------------------------------------------------------
__global__ __launch_bounds__(256) void qkv_gemm_kernel(
    const float* Xq, const float* Xk, const float* Xv,
    const __bf16* WqT, const __bf16* WkT, const __bf16* WvT,
    const float* bq, const float* bk, const float* bv,
    __bf16* Qb, __bf16* Kb, __bf16* Vtb) {
  const int mode = blockIdx.z;
  const float* X = (mode == 0) ? Xq : (mode == 1) ? Xk : Xv;
  const __bf16* Wt = (mode == 0) ? WqT : (mode == 1) ? WkT : WvT;
  const float* bias = (mode == 0) ? bq : (mode == 1) ? bk : bv;
  __bf16* Y = (mode == 0) ? Qb : (mode == 1) ? Kb : Vtb;

  const int tid = threadIdx.x, lane = tid & 31, wave = tid >> 5;
  const int lane16 = lane & 15, half = lane >> 4;
  const int wm = wave >> 2, wn = wave & 3;           // 2x4 wave grid
  const int rowBase = blockIdx.x * 64;
  const int colBase = blockIdx.y * 256;

  __shared__ __bf16 lA[64 * 40];       // 64x32, stride 40 (80B rows, 16B aligned)
  __shared__ __bf16 lB[2][256 * 40];   // double-buffered 256x32

  const __bf16* wrow = Wt + (size_t)(colBase + tid) * D_MODEL;

  // prologue: pre-issue B tile 0
  {
    __bf16* d = &lB[0][tid * 40];
    async_copy16(wrow,      d);
    async_copy16(wrow + 8,  d + 8);
    async_copy16(wrow + 16, d + 16);
    async_copy16(wrow + 24, d + 24);
  }

  v8f acc[2][4] = {};

  for (int k0 = 0; k0 < D_MODEL; k0 += 32) {
    const int cur = (k0 >> 5) & 1;
    {  // stage A (fp32 -> bf16 through VGPRs), 8 elems / thread
      int r = tid >> 2, cg = (tid & 3) * 8;
      const float* s = X + (size_t)(rowBase + r) * D_MODEL + k0 + cg;
      float4 f0 = *(const float4*)s;
      float4 f1 = *(const float4*)(s + 4);
      bf16x8 hh;
      hh[0] = (__bf16)f0.x; hh[1] = (__bf16)f0.y; hh[2] = (__bf16)f0.z; hh[3] = (__bf16)f0.w;
      hh[4] = (__bf16)f1.x; hh[5] = (__bf16)f1.y; hh[6] = (__bf16)f1.z; hh[7] = (__bf16)f1.w;
      *(bf16x8*)&lA[r * 40 + cg] = hh;
    }
    if (k0 + 32 < D_MODEL) {  // issue next B tile, wait only for current tile
      const __bf16* s = wrow + k0 + 32;
      __bf16* d = &lB[cur ^ 1][tid * 40];
      async_copy16(s,      d);
      async_copy16(s + 8,  d + 8);
      async_copy16(s + 16, d + 16);
      async_copy16(s + 24, d + 24);
      async_wait4();
    } else {
      async_wait0();
    }
    __syncthreads();
    // load ALL fragments first (simultaneously live -> distinct VGPRs,
    // lets the compiler stage s_wait_dscnt and overlap WMMAs with DS loads)
    const __bf16* bbase = &lB[cur][(wn * 64 + lane16) * 40];
    v16bf a0 = ld_frag_a(&lA[(wm * 32 + 0  + lane16) * 40], half);
    v16bf a1 = ld_frag_a(&lA[(wm * 32 + 16 + lane16) * 40], half);
    v16bf b0 = ld_frag_b(bbase + 0 * 16 * 40, half);
    v16bf b1 = ld_frag_b(bbase + 1 * 16 * 40, half);
    v16bf b2 = ld_frag_b(bbase + 2 * 16 * 40, half);
    v16bf b3 = ld_frag_b(bbase + 3 * 16 * 40, half);
    acc[0][0] = wmma_bf16(a0, b0, acc[0][0]);
    acc[1][0] = wmma_bf16(a1, b0, acc[1][0]);
    acc[0][1] = wmma_bf16(a0, b1, acc[0][1]);
    acc[1][1] = wmma_bf16(a1, b1, acc[1][1]);
    acc[0][2] = wmma_bf16(a0, b2, acc[0][2]);
    acc[1][2] = wmma_bf16(a1, b2, acc[1][2]);
    acc[0][3] = wmma_bf16(a0, b3, acc[0][3]);
    acc[1][3] = wmma_bf16(a1, b3, acc[1][3]);
    __syncthreads();
  }

#pragma unroll
  for (int i = 0; i < 2; ++i)
#pragma unroll
    for (int j = 0; j < 4; ++j)
#pragma unroll
      for (int r = 0; r < 8; ++r) {
        int mrow = rowBase + wm * 32 + i * 16 + r + 8 * half;
        int col  = colBase + wn * 64 + j * 16 + lane16;
        float v = acc[i][j][r] + bias[col];
        int b = mrow >> 11, l = mrow & (SEQ - 1);
        int h = col >> 6,  dh = col & (HEAD_DIM - 1);
        size_t off = (mode == 2)
            ? ((size_t)(b * N_HEADS + h) * HEAD_DIM + dh) * SEQ + l
            : ((size_t)(b * N_HEADS + h) * SEQ + l) * HEAD_DIM + dh;
        Y[off] = (__bf16)v;
      }
}

// ---------------------------------------------------------------------------
// Kernel 3: flash attention. grid = (QL/128, H, B); 8 waves x 16 queries.
// ---------------------------------------------------------------------------
__global__ __launch_bounds__(256) void attn_kernel(
    const __bf16* Qb, const __bf16* Kb, const __bf16* Vt, __bf16* Ob) {
  const int b = blockIdx.z, h = blockIdx.y;
  const int tid = threadIdx.x, lane = tid & 31, wave = tid >> 5;
  const int lane16 = lane & 15, half = lane >> 4;
  const int qbase = blockIdx.x * 128 + wave * 16;

  const __bf16* Qh = Qb + (size_t)(b * N_HEADS + h) * SEQ * HEAD_DIM;
  const __bf16* Kh = Kb + (size_t)(b * N_HEADS + h) * SEQ * HEAD_DIM;
  const __bf16* Vh = Vt + (size_t)(b * N_HEADS + h) * HEAD_DIM * SEQ;

  __shared__ __bf16 lP[8][16 * 80];   // per-wave 16x64 P tile, stride 80
  __bf16* pb = &lP[wave][0];

  // preload Q fragments (16 queries x 64 dim -> 2 A-frags)
  const __bf16* qrow = Qh + (size_t)(qbase + lane16) * HEAD_DIM;
  v16bf aq0 = ld_frag_a(qrow, half);
  v16bf aq1 = ld_frag_a(qrow + 32, half);

  v8f o[4] = {};
  float m[8], l[8];
#pragma unroll
  for (int r = 0; r < 8; ++r) { m[r] = -1e30f; l[r] = 0.0f; }

  for (int kt = 0; kt < SEQ / 64; ++kt) {
    // --- scores S = (Q K^T) * scale : 4 tiles of 16 keys --------------------
    v8f s[4];
#pragma unroll
    for (int j = 0; j < 4; ++j) {
      const __bf16* kp = Kh + (size_t)(kt * 64 + j * 16 + lane16) * HEAD_DIM;
      __builtin_prefetch(kp + 64 * HEAD_DIM, 0, 0);   // next K tile (L2-resident)
      v16bf bk0 = ld_frag_b(kp, half);        // d = 0..31
      v16bf bk1 = ld_frag_b(kp + 32, half);   // d = 32..63
      v8f sj = {};
      sj = wmma_bf16(aq0, bk0, sj);
      sj = wmma_bf16(aq1, bk1, sj);
      s[j] = sj * 0.125f;                     // 1/sqrt(64)
    }
    // --- online softmax: row max / rescale / row sum ------------------------
    float t[8];
#pragma unroll
    for (int r = 0; r < 8; ++r) {
      t[r] = fmaxf(fmaxf(s[0][r], s[1][r]), fmaxf(s[2][r], s[3][r]));
#pragma unroll
      for (int off = 1; off < 16; off <<= 1)
        t[r] = fmaxf(t[r], __shfl_xor(t[r], off, 32));
    }
    float corr[8];
#pragma unroll
    for (int r = 0; r < 8; ++r) {
      float mn = fmaxf(m[r], t[r]);
      corr[r] = __expf(m[r] - mn);
      m[r] = mn;
#pragma unroll
      for (int j = 0; j < 4; ++j) s[j][r] = __expf(s[j][r] - mn);
      float rs = s[0][r] + s[1][r] + s[2][r] + s[3][r];
#pragma unroll
      for (int off = 1; off < 16; off <<= 1) rs += __shfl_xor(rs, off, 32);
      l[r] = l[r] * corr[r] + rs;
#pragma unroll
      for (int jd = 0; jd < 4; ++jd) o[jd][r] *= corr[r];
    }
    // --- P: C-layout -> A-fragment layout via per-wave LDS ------------------
#pragma unroll
    for (int j = 0; j < 4; ++j)
#pragma unroll
      for (int r = 0; r < 8; ++r)
        pb[(r + 8 * half) * 80 + j * 16 + lane16] = (__bf16)s[j][r];
    __syncthreads();
    const __bf16* prow = pb + lane16 * 80;
    v16bf ap0 = ld_frag_a(prow, half);        // keys 0..31 of tile
    v16bf ap1 = ld_frag_a(prow + 32, half);   // keys 32..63
    // --- O += P V -----------------------------------------------------------
#pragma unroll
    for (int jd = 0; jd < 4; ++jd) {
      const __bf16* vp = Vh + (size_t)(jd * 16 + lane16) * SEQ + kt * 64;
      __builtin_prefetch(vp + 64, 0, 0);      // next V tile
      v16bf bv0 = ld_frag_b(vp, half);
      v16bf bv1 = ld_frag_b(vp + 32, half);
      o[jd] = wmma_bf16(ap0, bv0, o[jd]);
      o[jd] = wmma_bf16(ap1, bv1, o[jd]);
    }
    __syncthreads();
  }

  // epilogue: normalize, merge heads into Ob[b][l][h*64+d]
#pragma unroll
  for (int r = 0; r < 8; ++r) {
    float inv = 1.0f / l[r];
    int q = qbase + r + 8 * half;
#pragma unroll
    for (int jd = 0; jd < 4; ++jd) {
      int d = jd * 16 + lane16;
      Ob[((size_t)b * SEQ + q) * D_MODEL + h * HEAD_DIM + d] = (__bf16)(o[jd][r] * inv);
    }
  }
}

// ---------------------------------------------------------------------------
// Kernel 4: output projection. Out = Ob @ Wo + bo, fp32 output.
// A and B tiles both double-buffered via async global->LDS copies.
// ---------------------------------------------------------------------------
__global__ __launch_bounds__(256) void out_gemm_kernel(
    const __bf16* Xb, const __bf16* WoT, const float* bo, float* Out) {
  const int tid = threadIdx.x, lane = tid & 31, wave = tid >> 5;
  const int lane16 = lane & 15, half = lane >> 4;
  const int wm = wave >> 2, wn = wave & 3;
  const int rowBase = blockIdx.x * 64;
  const int colBase = blockIdx.y * 256;

  __shared__ __bf16 lA[2][64 * 40];
  __shared__ __bf16 lB[2][256 * 40];

  const int ar = tid >> 2, acg = (tid & 3) * 8;
  const __bf16* xrow = Xb + (size_t)(rowBase + ar) * D_MODEL + acg;
  const __bf16* wrow = WoT + (size_t)(colBase + tid) * D_MODEL;

  // prologue: pre-issue tile 0 (1 A copy + 4 B copies per thread)
  {
    async_copy16(xrow, &lA[0][ar * 40 + acg]);
    __bf16* d = &lB[0][tid * 40];
    async_copy16(wrow,      d);
    async_copy16(wrow + 8,  d + 8);
    async_copy16(wrow + 16, d + 16);
    async_copy16(wrow + 24, d + 24);
  }

  v8f acc[2][4] = {};

  for (int k0 = 0; k0 < D_MODEL; k0 += 32) {
    const int cur = (k0 >> 5) & 1;
    if (k0 + 32 < D_MODEL) {  // issue next tile, wait only for current tile
      async_copy16(xrow + k0 + 32, &lA[cur ^ 1][ar * 40 + acg]);
      const __bf16* s = wrow + k0 + 32;
      __bf16* d = &lB[cur ^ 1][tid * 40];
      async_copy16(s,      d);
      async_copy16(s + 8,  d + 8);
      async_copy16(s + 16, d + 16);
      async_copy16(s + 24, d + 24);
      async_wait5();
    } else {
      async_wait0();
    }
    __syncthreads();
    const __bf16* bbase = &lB[cur][(wn * 64 + lane16) * 40];
    v16bf a0 = ld_frag_a(&lA[cur][(wm * 32 + 0  + lane16) * 40], half);
    v16bf a1 = ld_frag_a(&lA[cur][(wm * 32 + 16 + lane16) * 40], half);
    v16bf b0 = ld_frag_b(bbase + 0 * 16 * 40, half);
    v16bf b1 = ld_frag_b(bbase + 1 * 16 * 40, half);
    v16bf b2 = ld_frag_b(bbase + 2 * 16 * 40, half);
    v16bf b3 = ld_frag_b(bbase + 3 * 16 * 40, half);
    acc[0][0] = wmma_bf16(a0, b0, acc[0][0]);
    acc[1][0] = wmma_bf16(a1, b0, acc[1][0]);
    acc[0][1] = wmma_bf16(a0, b1, acc[0][1]);
    acc[1][1] = wmma_bf16(a1, b1, acc[1][1]);
    acc[0][2] = wmma_bf16(a0, b2, acc[0][2]);
    acc[1][2] = wmma_bf16(a1, b2, acc[1][2]);
    acc[0][3] = wmma_bf16(a0, b3, acc[0][3]);
    acc[1][3] = wmma_bf16(a1, b3, acc[1][3]);
    __syncthreads();
  }

#pragma unroll
  for (int i = 0; i < 2; ++i)
#pragma unroll
    for (int j = 0; j < 4; ++j)
#pragma unroll
      for (int r = 0; r < 8; ++r) {
        int mrow = rowBase + wm * 32 + i * 16 + r + 8 * half;
        int col  = colBase + wn * 64 + j * 16 + lane16;
        Out[(size_t)mrow * D_MODEL + col] = acc[i][j][r] + bo[col];
      }
}

// ---------------------------------------------------------------------------
extern "C" void kernel_launch(void* const* d_in, const int* in_sizes, int n_in,
                              void* d_out, int out_size, void* d_ws, size_t ws_size,
                              hipStream_t stream) {
  const float* query = (const float*)d_in[0];
  const float* key   = (const float*)d_in[1];
  const float* value = (const float*)d_in[2];
  const float* Wq = (const float*)d_in[3];
  const float* bq = (const float*)d_in[4];
  const float* Wk = (const float*)d_in[5];
  const float* bk = (const float*)d_in[6];
  const float* Wv = (const float*)d_in[7];
  const float* bv = (const float*)d_in[8];
  const float* Wo = (const float*)d_in[9];
  const float* bo = (const float*)d_in[10];
  float* out = (float*)d_out;

  const size_t MB = (size_t)1 << 20;
  char* ws = (char*)d_ws;
  __bf16* WqT = (__bf16*)(ws + 0 * MB);
  __bf16* WkT = (__bf16*)(ws + 2 * MB);
  __bf16* WvT = (__bf16*)(ws + 4 * MB);
  __bf16* WoT = (__bf16*)(ws + 6 * MB);
  __bf16* Qb  = (__bf16*)(ws + 8 * MB);   // [B][H][L][Dh]
  __bf16* Kb  = (__bf16*)(ws + 16 * MB);  // [B][H][L][Dh]
  __bf16* Vtb = (__bf16*)(ws + 24 * MB);  // [B][H][Dh][L]
  __bf16* Ob  = (__bf16*)(ws + 32 * MB);  // [B][L][D]

  wtrans_kernel<<<dim3(32, 32, 4), dim3(32, 8), 0, stream>>>(
      Wq, Wk, Wv, Wo, WqT, WkT, WvT, WoT);

  qkv_gemm_kernel<<<dim3(64, 4, 3), 256, 0, stream>>>(
      query, key, value, WqT, WkT, WvT, bq, bk, bv, Qb, Kb, Vtb);

  attn_kernel<<<dim3(SEQ / 128, N_HEADS, BATCH), 256, 0, stream>>>(
      Qb, Kb, Vtb, Ob);

  out_gemm_kernel<<<dim3(64, 4, 1), 256, 0, stream>>>(Ob, WoT, bo, out);
}
